// TwoDirectionalMultiDimensionalAttention_65214783422466
// MI455X (gfx1250) — compile-verified, matrix operand order checked by reading, not run on hardware
//
#include <hip/hip_runtime.h>
#include <hip/hip_bf16.h>
#include <cstdint>

// ---------------------------------------------------------------------------
// Problem constants (B=4, S=2048, E=1024, ND=2)
// ---------------------------------------------------------------------------
#define B_   4
#define S_   2048
#define E_   1024
#define ND_  2
#define NTOK (B_ * S_)        // 8192 tokens
#define F2   (ND_ * E_)       // 2048 features per projection
#define F6   (6 * F2)         // 12288 = all six projections stacked

typedef __attribute__((ext_vector_type(16))) __bf16 v16bf;
typedef __attribute__((ext_vector_type(8)))  __bf16 v8bf;
typedef __attribute__((ext_vector_type(8)))  float  v8f;

// ---------------------------------------------------------------------------
// Tensor Data Mover availability / arity detection (see CDNA5_HIP.md):
//   ROCm 7.2 (clang-22):      5-arg __builtin_amdgcn_tensor_load_to_lds
//   amdgpu-toolchain (cl-23): 6-arg form; ships amd_gfx1250_TDM.h
// ---------------------------------------------------------------------------
#if defined(__has_builtin)
#if __has_builtin(__builtin_amdgcn_tensor_load_to_lds)
#define HAVE_TDM 1
#endif
#endif
#ifndef HAVE_TDM
#define HAVE_TDM 0
#endif
#if defined(__has_include)
#if __has_include(<hip/amd_detail/amd_gfx1250_TDM.h>)
#define HAVE_TDM6 1
#endif
#endif
#ifndef HAVE_TDM6
#define HAVE_TDM6 0
#endif

#if HAVE_TDM
typedef __attribute__((ext_vector_type(4))) unsigned int u32x4;
typedef __attribute__((ext_vector_type(8))) int          i32x8;
typedef __attribute__((ext_vector_type(4))) int          i32x4;

// Issue one TDM descriptor: load a 128-row x 32-col bf16 tile (row stride
// lda elements in global) into LDS at lds_dst, writing rows of 64B with 16B
// padding after each (pad_interval=3 -> every 2^3*8=64B, pad_amount=3 ->
// 4 DWORDs) so the LDS layout matches the 40-element row stride used by the
// WMMA fragment loader. D# packing per CDNA5 ISA ch.8 (async_tensor).
__device__ __forceinline__ void tdm_load_tile_128x32(const __bf16* gsrc,
                                                     long lda,
                                                     const __bf16* lds_dst) {
  const unsigned long long ga = (unsigned long long)(uintptr_t)gsrc;
  const unsigned int ldsa = (unsigned int)(uintptr_t)lds_dst;  // addr[31:0]=LDS offset
  const unsigned int td0 = 1u << 20;   // generous tensor dims (OOB never hit;
  const unsigned int td1 = 1u << 20;   // all shapes are tile-aligned)
  const unsigned int tile0 = 32, tile1 = 128;
  const unsigned long long str0 = (unsigned long long)lda;  // elements

  u32x4 g0;
  g0[0] = 1u;                                   // count=1, user descriptor
  g0[1] = ldsa;                                 // lds_addr
  g0[2] = (unsigned int)ga;                     // global_addr[31:0]
  g0[3] = (unsigned int)((ga >> 32) & 0x1FFFFFFull) | (2u << 30);  // [56:32]|type=2

  i32x8 g1;
  g1[0] = (int)((1u << 16)        // data_size = 2 bytes
              | (1u << 20)        // pad_enable
              | (3u << 22)        // pad_interval: 64B
              | (3u << 25));      // pad_amount: 16B
  g1[1] = (int)((td0 & 0xFFFFu) << 16);                           // dim0 lo
  g1[2] = (int)(((td0 >> 16) & 0xFFFFu) | ((td1 & 0xFFFFu) << 16));
  g1[3] = (int)(((td1 >> 16) & 0xFFFFu) | (tile0 << 16));
  g1[4] = (int)(tile1);                                           // tile_dim1, tile_dim2=0
  g1[5] = (int)(unsigned int)(str0 & 0xFFFFFFFFull);              // stride0 lo
  g1[6] = (int)(unsigned int)((str0 >> 32) & 0xFFFFull);          // stride0 hi
  g1[7] = 0;

  i32x4 z4 = {0, 0, 0, 0};
#if HAVE_TDM6
  i32x8 z8 = {0, 0, 0, 0, 0, 0, 0, 0};
  __builtin_amdgcn_tensor_load_to_lds(g0, g1, z4, z4, z8, 0);
#else
  __builtin_amdgcn_tensor_load_to_lds(g0, g1, z4, z4, 0);
#endif
}
#endif  // HAVE_TDM

// ---------------------------------------------------------------------------
// Elementwise helpers
// ---------------------------------------------------------------------------
__global__ void k_cvt_f32_bf16(const float* __restrict__ in,
                               __bf16* __restrict__ out, int n) {
  int i = blockIdx.x * blockDim.x + threadIdx.x;
  int stride = gridDim.x * blockDim.x;
  for (; i < n; i += stride) out[i] = (__bf16)in[i];
}

__global__ void k_zero_f32(float* __restrict__ p, int n) {
  int i = blockIdx.x * blockDim.x + threadIdx.x;
  int stride = gridDim.x * blockDim.x;
  for (; i < n; i += stride) p[i] = 0.0f;
}

__global__ void k_pack_bias(const float* __restrict__ b0, const float* __restrict__ b1,
                            const float* __restrict__ b2, const float* __restrict__ b3,
                            const float* __restrict__ b4, const float* __restrict__ b5,
                            float* __restrict__ out) {
  int i = blockIdx.x * blockDim.x + threadIdx.x;   // n = F6
  if (i >= F6) return;
  const float* bs[6] = {b0, b1, b2, b3, b4, b5};
  out[i] = bs[i >> 11][i & (F2 - 1)];
}

// ---------------------------------------------------------------------------
// WMMA fragment loader (16-bit A 16x32 layout, ISA 7.12.2): lane L holds row
// M = L%16; lanes 0-15: K={0..7,16..23}; lanes 16-31: K={8..15,24..31}.
// Two contiguous 16B LDS loads per lane from a row-major tile. The B-fragment
// of B[k][n] uses the same loader on a tile stored row-major by n (B^T).
// ---------------------------------------------------------------------------
__device__ __forceinline__ v16bf load_frag(const __bf16* base, int stride) {
  const int lane = threadIdx.x & 31;
  const int row  = lane & 15;
  const int g    = lane >> 4;
  const __bf16* p = base + row * stride + (g << 3);
  v8bf lo = *(const v8bf*)(p);
  v8bf hi = *(const v8bf*)(p + 16);
  return __builtin_shufflevector(lo, hi, 0, 1, 2, 3, 4, 5, 6, 7,
                                         8, 9, 10, 11, 12, 13, 14, 15);
}

enum { MODE_PROJ = 0, MODE_SCORES = 1, MODE_ADD = 2, MODE_OUT = 3 };

template <int MODE>
__device__ __forceinline__ void epilogue(v8f (&acc)[4][2], void* Cv, long ldc,
                                         const float* epi, float alpha,
                                         long row0, long col0) {
  const int tid  = threadIdx.x;
  const int wid  = tid >> 5;
  const int lane = tid & 31;
  const int wm = wid >> 2, wn = wid & 3;
  const int lrow = lane & 15, g = lane >> 4;
#pragma unroll
  for (int mi = 0; mi < 4; mi++) {
#pragma unroll
    for (int ni = 0; ni < 2; ni++) {
#pragma unroll
      for (int r = 0; r < 8; r++) {
        const long gm = row0 + wm * 64 + mi * 16 + r + 8 * g;
        const long gn = col0 + wn * 32 + ni * 16 + lrow;
        const float v = acc[mi][ni][r];
        if constexpr (MODE == MODE_PROJ) {
          ((__bf16*)Cv)[gm * ldc + gn] = (__bf16)(v + epi[gn]);
        } else if constexpr (MODE == MODE_SCORES) {
          ((float*)Cv)[gm * ldc + gn] = v * alpha * epi[gn];
        } else if constexpr (MODE == MODE_ADD) {
          // Fire-and-forget L2 f32 atomic (global_atomic_add_f32, no return):
          // keeps the accumulation off the latency critical path and makes
          // the two direction launches order-independent.
          unsafeAtomicAdd(&((float*)Cv)[gm * ldc + gn], v);
        } else {
          ((float*)Cv)[gm * ldc + gn] = v + epi[gn];
        }
      }
    }
  }
}

__device__ __forceinline__ void wmma_tile(v8f (&acc)[4][2], const __bf16* As,
                                          const __bf16* Bs) {
  const int wid = threadIdx.x >> 5;
  const int wm = wid >> 2, wn = wid & 3;
  v16bf af[4], bf[2];
#pragma unroll
  for (int mi = 0; mi < 4; mi++)
    af[mi] = load_frag(&As[(wm * 64 + mi * 16) * 40], 40);
#pragma unroll
  for (int ni = 0; ni < 2; ni++)
    bf[ni] = load_frag(&Bs[(wn * 32 + ni * 16) * 40], 40);
#pragma unroll
  for (int mi = 0; mi < 4; mi++)
#pragma unroll
    for (int ni = 0; ni < 2; ni++)
      acc[mi][ni] = __builtin_amdgcn_wmma_f32_16x16x32_bf16(
          false, af[mi], false, bf[ni], (short)0, acc[mi][ni], false, false);
}

// ---------------------------------------------------------------------------
// C = A @ B^T   (A: MxK row-major, B: NxK row-major), 128x128 tile per
// 256-thread / 8-wave workgroup, BK=32, waves in 2(m)x4(n) grid, 4x2 16x16
// subtiles per wave -> 8 v_wmma per K-step.
//
// With TDM: wave 0 drives double-buffered DMA tile loads (TENSORcnt),
// everyone else just computes. Without TDM: synchronous vector staging.
// ---------------------------------------------------------------------------
template <int MODE>
__global__ __launch_bounds__(256) void k_gemm_nt(
    const __bf16* __restrict__ A, long lda,
    const __bf16* __restrict__ Bm, long ldb,
    void* __restrict__ Cv, long ldc,
    const float* __restrict__ epi, float alpha, int K) {
  const long row0 = (long)blockIdx.y * 128;
  const long col0 = (long)blockIdx.x * 128;
  v8f acc[4][2] = {};

#if HAVE_TDM
  __shared__ alignas(16) __bf16 As[2][128 * 40];
  __shared__ alignas(16) __bf16 Bs[2][128 * 40];
  const int wid = threadIdx.x >> 5;
  const int nsteps = K / 32;
  if (wid == 0) {
    tdm_load_tile_128x32(A + row0 * lda, lda, As[0]);
    tdm_load_tile_128x32(Bm + col0 * ldb, ldb, Bs[0]);
  }
  for (int step = 0; step < nsteps; ++step) {
    const int cur = step & 1;
    if (wid == 0) {
      if (step + 1 < nsteps) {
        tdm_load_tile_128x32(A + row0 * lda + (step + 1) * 32, lda, As[cur ^ 1]);
        tdm_load_tile_128x32(Bm + col0 * ldb + (step + 1) * 32, ldb, Bs[cur ^ 1]);
        __builtin_amdgcn_s_wait_tensorcnt(2);  // current pair has landed
      } else {
        __builtin_amdgcn_s_wait_tensorcnt(0);
      }
    }
    __syncthreads();                 // publish TDM-written LDS to all waves
    wmma_tile(acc, As[cur], Bs[cur]);
    __syncthreads();                 // LDS reads done before next DMA overwrite
  }
#else
  __shared__ alignas(16) __bf16 As[128 * 40];
  __shared__ alignas(16) __bf16 Bs[128 * 40];
  const int tid = threadIdx.x;
  const int arow = tid >> 1;
  const int acol = (tid & 1) * 16;
  for (int k0 = 0; k0 < K; k0 += 32) {
    const uint4* ap = (const uint4*)(A + (row0 + arow) * lda + k0 + acol);
    uint4 av0 = ap[0], av1 = ap[1];
    const uint4* bp = (const uint4*)(Bm + (col0 + arow) * ldb + k0 + acol);
    uint4 bv0 = bp[0], bv1 = bp[1];
    __syncthreads();
    *(uint4*)&As[arow * 40 + acol]     = av0;
    *(uint4*)&As[arow * 40 + acol + 8] = av1;
    *(uint4*)&Bs[arow * 40 + acol]     = bv0;
    *(uint4*)&Bs[arow * 40 + acol + 8] = bv1;
    __syncthreads();
    wmma_tile(acc, As, Bs);
  }
#endif
  epilogue<MODE>(acc, Cv, ldc, epi, alpha, row0, col0);
}

// ---------------------------------------------------------------------------
// C += A @ B    (A: MxK row-major bf16 probs, B: KxN row-major, ldb strided).
// B tile is transposed into LDS during staging (TDM cannot transpose), so
// fragment loads stay contiguous. Synchronous single-buffer staging;
// accumulation into C via f32 atomics.
// ---------------------------------------------------------------------------
__global__ __launch_bounds__(256) void k_gemm_av(
    const __bf16* __restrict__ A, long lda,
    const __bf16* __restrict__ Bm, long ldb,
    float* __restrict__ C, long ldc, int K) {
  __shared__ alignas(16) __bf16 As[128 * 40];
  __shared__ alignas(16) __bf16 Bs[128 * 40];
  const int  tid  = threadIdx.x;
  const long row0 = (long)blockIdx.y * 128;
  const long col0 = (long)blockIdx.x * 128;
  const int arow = tid >> 1;
  const int acol = (tid & 1) * 16;
  const int bk = tid >> 3;          // 0..31 (k row)
  const int bn = (tid & 7) * 16;    // 0..112 (n col)
  v8f acc[4][2] = {};

  for (int k0 = 0; k0 < K; k0 += 32) {
    const uint4* ap = (const uint4*)(A + (row0 + arow) * lda + k0 + acol);
    uint4 av0 = ap[0], av1 = ap[1];
    const uint4* bp = (const uint4*)(Bm + (long)(k0 + bk) * ldb + col0 + bn);
    uint4 bv0 = bp[0], bv1 = bp[1];
    __syncthreads();
    *(uint4*)&As[arow * 40 + acol]     = av0;
    *(uint4*)&As[arow * 40 + acol + 8] = av1;
    {
      alignas(16) __bf16 tmp[16];
      *(uint4*)(tmp)     = bv0;
      *(uint4*)(tmp + 8) = bv1;
#pragma unroll
      for (int j = 0; j < 16; j++) Bs[(bn + j) * 40 + bk] = tmp[j];
    }
    __syncthreads();
    wmma_tile(acc, As, Bs);
  }
  epilogue<MODE_ADD>(acc, (void*)C, ldc, nullptr, 1.0f, row0, col0);
}

// ---------------------------------------------------------------------------
// Row softmax: one 256-thread block per row of S_ f32 scores (already scaled
// and tfidf-weighted); writes bf16 probabilities.
// ---------------------------------------------------------------------------
__global__ __launch_bounds__(256) void k_softmax(const float* __restrict__ Sc,
                                                 __bf16* __restrict__ P) {
  const int row = blockIdx.x;
  const float* src = Sc + (size_t)row * S_;
  __bf16* dst = P + (size_t)row * S_;
  __shared__ float red[256];
  const int tid = threadIdx.x;

  float m = -3.4e38f;
  for (int c = tid; c < S_; c += 256) m = fmaxf(m, src[c]);
  red[tid] = m;
  __syncthreads();
  for (int s = 128; s > 0; s >>= 1) {
    if (tid < s) red[tid] = fmaxf(red[tid], red[tid + s]);
    __syncthreads();
  }
  m = red[0];
  __syncthreads();

  float sum = 0.0f;
  for (int c = tid; c < S_; c += 256) sum += __expf(src[c] - m);
  red[tid] = sum;
  __syncthreads();
  for (int s = 128; s > 0; s >>= 1) {
    if (tid < s) red[tid] += red[tid + s];
    __syncthreads();
  }
  const float inv = 1.0f / red[0];

  for (int c = tid; c < S_; c += 256)
    dst[c] = (__bf16)(__expf(src[c] - m) * inv);
}

// ---------------------------------------------------------------------------
// Host-side orchestration. The reference's sequence reversals cancel exactly
// (softmax is permutation-invariant over keys), so both directions run in
// forward index order with their own weights.
// ---------------------------------------------------------------------------
extern "C" void kernel_launch(void* const* d_in, const int* in_sizes, int n_in,
                              void* d_out, int out_size, void* d_ws,
                              size_t ws_size, hipStream_t stream) {
  (void)in_sizes; (void)n_in; (void)out_size; (void)ws_size;

  const float* x     = (const float*)d_in[0];
  const float* tfidf = (const float*)d_in[1];
  const float* W[6]  = {(const float*)d_in[2],  (const float*)d_in[4],
                        (const float*)d_in[6],  (const float*)d_in[8],
                        (const float*)d_in[10], (const float*)d_in[12]};
  const float* bias[6] = {(const float*)d_in[3],  (const float*)d_in[5],
                          (const float*)d_in[7],  (const float*)d_in[9],
                          (const float*)d_in[11], (const float*)d_in[13]};
  const float* Wo = (const float*)d_in[14];
  const float* bo = (const float*)d_in[15];
  float* out = (float*)d_out;

  char* w = (char*)d_ws;
  size_t off = 0;
  auto carve = [&](size_t bytes) {
    void* p = w + off;
    off = (off + bytes + 255) & ~(size_t)255;
    return p;
  };
  __bf16* xh    = (__bf16*)carve((size_t)NTOK * E_ * 2);    //  16 MB
  __bf16* W6h   = (__bf16*)carve((size_t)F6 * E_ * 2);      //  24 MB
  __bf16* Woh   = (__bf16*)carve((size_t)E_ * F2 * 2);      //   4 MB
  float*  bias6 = (float*)carve((size_t)F6 * 4);            //  48 KB
  __bf16* Y     = (__bf16*)carve((size_t)NTOK * F6 * 2);    // 192 MB QKV(all)
  float*  Sc    = (float*)carve((size_t)S_ * S_ * 4);       //  16 MB
  __bf16* Pb    = (__bf16*)carve((size_t)S_ * S_ * 2);      //   8 MB
  float*  comb  = (float*)carve((size_t)NTOK * F2 * 4);     //  64 MB
  __bf16* combh = (__bf16*)carve((size_t)NTOK * F2 * 2);    //  32 MB

  const float scale = 1.0f / 32.0f;  // 1/sqrt(E)

  // 1) bf16 conversions
  k_cvt_f32_bf16<<<2048, 256, 0, stream>>>(x, xh, NTOK * E_);
  for (int p = 0; p < 6; p++)
    k_cvt_f32_bf16<<<1024, 256, 0, stream>>>(W[p], W6h + (size_t)p * F2 * E_,
                                             F2 * E_);
  k_cvt_f32_bf16<<<1024, 256, 0, stream>>>(Wo, Woh, E_ * F2);
  k_pack_bias<<<F6 / 256, 256, 0, stream>>>(bias[0], bias[1], bias[2],
                                            bias[3], bias[4], bias[5], bias6);

  // 2) All six projections in one GEMM: Y[8192 x 12288] = xh @ W6h^T + bias6
  k_gemm_nt<MODE_PROJ><<<dim3(F6 / 128, NTOK / 128), 256, 0, stream>>>(
      xh, E_, W6h, E_, Y, F6, bias6, 1.0f, E_);

  // 3) Attention: 16 problems (b, dir, d): scores -> softmax -> P@V (atomic +=)
  k_zero_f32<<<2048, 256, 0, stream>>>(comb, NTOK * F2);
  for (int b = 0; b < B_; b++) {
    for (int dir = 0; dir < 2; dir++) {
      for (int d = 0; d < ND_; d++) {
        const size_t tokbase = (size_t)b * S_ * F6;
        const __bf16* Q  = Y + tokbase + (size_t)(dir * 3 + 0) * F2 + (size_t)d * E_;
        const __bf16* Kp = Y + tokbase + (size_t)(dir * 3 + 1) * F2 + (size_t)d * E_;
        const __bf16* V  = Y + tokbase + (size_t)(dir * 3 + 2) * F2 + (size_t)d * E_;

        // Sc[s,t] = (Q@K^T)[s,t] * scale * tfidf[b,t]
        k_gemm_nt<MODE_SCORES><<<dim3(S_ / 128, S_ / 128), 256, 0, stream>>>(
            Q, F6, Kp, F6, Sc, S_, tfidf + (size_t)b * S_, scale, E_);
        k_softmax<<<S_, 256, 0, stream>>>(Sc, Pb);
        // comb[b, s, d*E + e] += (Pb @ V)[s, e]   (dirs accumulate atomically)
        k_gemm_av<<<dim3(E_ / 128, S_ / 128), 256, 0, stream>>>(
            Pb, S_, V, F6, comb + (size_t)b * S_ * F2 + (size_t)d * E_, F2, S_);
      }
    }
  }

  // 4) Output projection: out = combh @ Wo^T + bo
  k_cvt_f32_bf16<<<2048, 256, 0, stream>>>(comb, combh, NTOK * F2);
  k_gemm_nt<MODE_OUT><<<dim3(E_ / 128, NTOK / 128), 256, 0, stream>>>(
      combh, F2, Woh, F2, out, E_, bo, 1.0f, F2);
}
